// SimilarityLoss_39316130627934
// MI455X (gfx1250) — compile-verified, matrix operand order checked by reading, not run on hardware
//
#include <hip/hip_runtime.h>
#include <hip/hip_bf16.h>

// ---------------------------------------------------------------------------
// SimilarityLoss for MI455X (gfx1250, wave32, WMMA)
//   loss = mean( (normalize(E) @ normalize(E)^T - S)^2 )
//   E:[8192,128] f32, S:[8192,8192] f32
//
// Roofline: S stream = 256 MB @ 23.3 TB/s ~= 11 us (floor). GEMM = 17.2 GFLOP;
// bf16 WMMA (~2.5 PFLOPS dense) ~= 7 us, hidden under the stream. fp32 WMMA
// (16x16x4) would be ~55 us => bf16 A/B + f32 accumulate is the chip-rewarded
// precision. S (256 MB > 192 MB L2, read once) is loaded non-temporal so the
// 2 MB L2-resident bf16 X (reused 512x per side) is never evicted.
// ---------------------------------------------------------------------------

typedef __attribute__((ext_vector_type(16))) __bf16 v16bf;
typedef __attribute__((ext_vector_type(8)))  __bf16 v8bf;
typedef __attribute__((ext_vector_type(4)))  __bf16 v4bf;
typedef __attribute__((ext_vector_type(8)))  float  v8f;

#define SL_N      8192
#define SL_D      128
#define SL_TILES  (SL_N / 16)      // 512 tiles per dimension
#define SL_JB     4                // column tiles per wave
#define SL_WAVES  8                // waves per block (256 threads)
#define SL_GEMM_BLOCKS ((SL_TILES * (SL_TILES / SL_JB)) / SL_WAVES)  // 8192
#define SL_EPS    1e-8f

// ---------------------------------------------------------------------------
// Kernel 1: row L2-normalize fp32 -> bf16.  One wave per row (4 floats/lane).
// ---------------------------------------------------------------------------
__global__ void sl_normalize_kernel(const float* __restrict__ emb,
                                    __bf16* __restrict__ xbf) {
    const int wave = threadIdx.x >> 5;
    const int lane = threadIdx.x & 31;
    const int row  = blockIdx.x * SL_WAVES + wave;

    const float4 v = *(const float4*)(emb + (size_t)row * SL_D + lane * 4);
    float ss = v.x * v.x + v.y * v.y + v.z * v.z + v.w * v.w;
    #pragma unroll
    for (int off = 16; off >= 1; off >>= 1)
        ss += __shfl_xor(ss, off, 32);

    const float scale = 1.0f / fmaxf(sqrtf(ss), SL_EPS);

    v4bf o;
    o[0] = (__bf16)(v.x * scale);
    o[1] = (__bf16)(v.y * scale);
    o[2] = (__bf16)(v.z * scale);
    o[3] = (__bf16)(v.w * scale);
    *(v4bf*)(xbf + (size_t)row * SL_D + lane * 4) = o;
}

// ---------------------------------------------------------------------------
// Fragment loader for v_wmma_f32_16x16x32_bf16, 16-bit A (and, by symmetry of
// C = X X^T, B) layout:
//   lanes 0-15 : M=lane,    VGPR0..3 = K[kb+0..7],  VGPR4..7 = K[kb+16..23]
//   lanes16-31 : M=lane-16, VGPR0..3 = K[kb+8..15], VGPR4..7 = K[kb+24..31]
// Row-major memory => two contiguous 16B loads per lane.
// ---------------------------------------------------------------------------
__device__ __forceinline__ v16bf sl_load_frag(const __bf16* __restrict__ rowp,
                                              int kbase, int half) {
    const v8bf lo = *(const v8bf*)(rowp + kbase + half * 8);
    const v8bf hi = *(const v8bf*)(rowp + kbase + 16 + half * 8);
    return __builtin_shufflevector(lo, hi, 0, 1, 2, 3, 4, 5, 6, 7,
                                           8, 9, 10, 11, 12, 13, 14, 15);
}

// ---------------------------------------------------------------------------
// Kernel 2: fused GEMM (bf16 WMMA, f32 accum) + squared-error vs streamed S.
// One wave handles one 16-row tile x SL_JB 16-col tiles.
// S is loaded non-temporal (read-once, > L2 capacity).
// Per-block partial sum -> d_ws (deterministic ordering, no float atomics).
// ---------------------------------------------------------------------------
__global__ void sl_gemm_loss_kernel(const __bf16* __restrict__ xbf,
                                    const float* __restrict__ sim,
                                    float* __restrict__ partials) {
    const int wave = threadIdx.x >> 5;
    const int lane = threadIdx.x & 31;
    const int half = lane >> 4;     // K sub-block select
    const int m    = lane & 15;     // row within tile (A/B), col within tile (C)

    const int job    = blockIdx.x * SL_WAVES + wave;     // [0, 512*128)
    const int i_tile = job >> 7;                         // [0, 512)
    const int j0t    = (job & 127) * SL_JB;              // [0, 512) step 4

    // A fragments: row (i_tile*16 + m), all of K=128 in 4 chunks of 32.
    const __bf16* arow = xbf + (size_t)(i_tile * 16 + m) * SL_D;
    v16bf afrag[4];
    #pragma unroll
    for (int kc = 0; kc < 4; ++kc)
        afrag[kc] = sl_load_frag(arow, kc * 32, half);

    float err = 0.0f;

    #pragma unroll
    for (int jj = 0; jj < SL_JB; ++jj) {
        const int j_tile = j0t + jj;
        const __bf16* brow = xbf + (size_t)(j_tile * 16 + m) * SL_D;

        v8f c = {};
        #pragma unroll
        for (int kc = 0; kc < 4; ++kc) {
            const v16bf bfrag = sl_load_frag(brow, kc * 32, half);
            c = __builtin_amdgcn_wmma_f32_16x16x32_bf16(
                    /*neg_a=*/false, afrag[kc],
                    /*neg_b=*/false, bfrag,
                    /*c_mod=*/(short)0, c,
                    /*reuse_a=*/false, /*reuse_b=*/false);
        }

        // C layout: VGPR r -> (M = half*8 + r, N = m)
        const int col = j_tile * 16 + m;
        #pragma unroll
        for (int r = 0; r < 8; ++r) {
            const int row = i_tile * 16 + half * 8 + r;
            // Non-temporal: S is streamed exactly once and exceeds L2 capacity.
            const float s =
                __builtin_nontemporal_load(sim + (size_t)row * SL_N + col);
            const float d = c[r] - s;
            err = fmaf(d, d, err);
        }
    }

    // Wave reduce (fixed order), then block reduce via LDS.
    #pragma unroll
    for (int off = 16; off >= 1; off >>= 1)
        err += __shfl_xor(err, off, 32);

    __shared__ float wsum[SL_WAVES];
    if (lane == 0) wsum[wave] = err;
    __syncthreads();
    if (threadIdx.x == 0) {
        float t = 0.0f;
        #pragma unroll
        for (int w = 0; w < SL_WAVES; ++w) t += wsum[w];
        partials[blockIdx.x] = t;
    }
}

// ---------------------------------------------------------------------------
// Kernel 3: deterministic final reduction of SL_GEMM_BLOCKS partials -> mean.
// ---------------------------------------------------------------------------
__global__ void sl_reduce_kernel(const float* __restrict__ partials,
                                 float* __restrict__ out) {
    __shared__ float smem[256];
    float s = 0.0f;
    for (int i = threadIdx.x; i < SL_GEMM_BLOCKS; i += 256)
        s += partials[i];
    smem[threadIdx.x] = s;
    __syncthreads();
    #pragma unroll
    for (int off = 128; off >= 1; off >>= 1) {
        if (threadIdx.x < off) smem[threadIdx.x] += smem[threadIdx.x + off];
        __syncthreads();
    }
    if (threadIdx.x == 0)
        out[0] = smem[0] * (1.0f / 67108864.0f);   // / 8192^2  (exact pow2)
}

// ---------------------------------------------------------------------------
extern "C" void kernel_launch(void* const* d_in, const int* in_sizes, int n_in,
                              void* d_out, int out_size, void* d_ws, size_t ws_size,
                              hipStream_t stream) {
    (void)in_sizes; (void)n_in; (void)out_size; (void)ws_size;

    const float* emb = (const float*)d_in[0];   // [8192,128] f32
    const float* sim = (const float*)d_in[1];   // [8192,8192] f32
    float* out = (float*)d_out;                 // scalar loss

    __bf16* xbf      = (__bf16*)d_ws;                                    // 2 MB
    float*  partials = (float*)((char*)d_ws + (size_t)SL_N * SL_D * 2);  // 32 KB

    // 1) normalize rows -> bf16
    sl_normalize_kernel<<<SL_N / SL_WAVES, 256, 0, stream>>>(emb, xbf);
    // 2) fused WMMA GEMM + squared error (streams S exactly once, NT)
    sl_gemm_loss_kernel<<<SL_GEMM_BLOCKS, 256, 0, stream>>>(xbf, sim, partials);
    // 3) deterministic mean
    sl_reduce_kernel<<<1, 256, 0, stream>>>(partials, out);
}
